// Model_68212670595891
// MI455X (gfx1250) — compile-verified, matrix operand order checked by reading, not run on hardware
//
#include <hip/hip_runtime.h>
#include <hip/hip_bf16.h>
#include <math.h>

typedef __bf16 bf16_t;
typedef __attribute__((ext_vector_type(16))) __bf16 v16bf;
typedef __attribute__((ext_vector_type(8)))  float  v8f;

#define T_SEQ 70
#define B_SZ  80
#define N_ROWS (T_SEQ * B_SZ)   /* 5600 */
#define V_SZ  33278
#define E_SZ  400
#define H_SZ  1150
#define EP_SZ 416                /* E padded to multiple of 32 */
#define HP_SZ 1152               /* H padded to multiple of 32 */

#if __has_builtin(__builtin_amdgcn_global_load_async_to_lds_b128) && \
    __has_builtin(__builtin_amdgcn_s_wait_asynccnt)
#define HAVE_ASYNC_LDS 1
typedef __attribute__((address_space(1))) void gvoid;
typedef __attribute__((address_space(3))) void lvoid;
typedef int v4i __attribute__((vector_size(16)));
typedef __attribute__((address_space(1))) v4i gv4i;
typedef __attribute__((address_space(3))) v4i lv4i;
#else
#define HAVE_ASYNC_LDS 0
#endif

// ---------------------------------------------------------------------------
// Pack weight W [NC=N, K] row-major fp32 into WMMA B-fragment layout (bf16):
// logical B = W^T, i.e. B[k][n] = W[n*K + k].
// BP[(nt*KT + kt)*512 + l*16 + j] = B[kt*32 + (l>>4)*16 + j][nt*16 + (l&15)]
// Zero-filled outside logical bounds (handles ragged K and N).
// ---------------------------------------------------------------------------
__global__ void k_pack_b_frag(const float* __restrict__ W, bf16_t* __restrict__ BP,
                              int NC, int K, int KT, long tot) {
    long idx = (long)blockIdx.x * blockDim.x + threadIdx.x;
    if (idx >= tot) return;
    int  j  = (int)(idx & 15);
    int  l  = (int)((idx >> 4) & 31);
    long q  = idx >> 9;                 // nt*KT + kt
    int  kt = (int)(q % KT);
    int  nt = (int)(q / KT);
    int  n  = nt * 16 + (l & 15);
    int  k  = kt * 32 + (l >> 4) * 16 + j;
    BP[idx] = (n < NC && k < K) ? (bf16_t)W[(long)n * K + k] : (bf16_t)0.0f;
}

__global__ void k_bias_sum(const float* __restrict__ a,
                           const float* __restrict__ b,
                           float* __restrict__ o, int n) {
    int i = blockIdx.x * blockDim.x + threadIdx.x;
    if (i < n) o[i] = a[i] + b[i];
}

// Embedding gather -> bf16 activations [N_ROWS, EP_SZ]; pad columns zeroed.
__global__ void k_embed(const int* __restrict__ tok,
                        const float* __restrict__ Wemb,
                        bf16_t* __restrict__ x0) {
    long i = (long)blockIdx.x * blockDim.x + threadIdx.x;
    long tot = (long)N_ROWS * EP_SZ;
    if (i >= tot) return;
    int row = (int)(i / EP_SZ);
    int e   = (int)(i % EP_SZ);
    x0[i] = (e < E_SZ) ? (bf16_t)Wemb[(long)tok[row] * E_SZ + e] : (bf16_t)0.0f;
}

// Zero pad columns [G, ldp) of a padded bf16 activation buffer.
__global__ void k_zero_pad(bf16_t* __restrict__ buf, int rows, int G, int ldp) {
    int pad = ldp - G;
    int tot = rows * pad;
    int i = blockIdx.x * blockDim.x + threadIdx.x;
    if (i >= tot) return;
    int r = i / pad, j = i % pad;
    buf[(long)r * ldp + G + j] = (bf16_t)0.0f;
}

// Initialize h,c state; also zero hB pad columns.
__global__ void k_init_state(const float* __restrict__ h0,
                             const float* __restrict__ c0,
                             float* __restrict__ hF,
                             bf16_t* __restrict__ hB,
                             float* __restrict__ cst, int G, int ldp) {
    int i = blockIdx.x * blockDim.x + threadIdx.x;
    if (i >= B_SZ * ldp) return;
    int b = i / ldp, j = i % ldp;
    if (j < G) {
        float h = h0[b * G + j];
        hF[b * G + j]  = h;
        cst[b * G + j] = c0[b * G + j];
        hB[i] = (bf16_t)h;
    } else {
        hB[i] = (bf16_t)0.0f;
    }
}

__global__ void k_copy2(const float* __restrict__ a, const float* __restrict__ b,
                        float* __restrict__ oa, float* __restrict__ ob, int n) {
    int i = blockIdx.x * blockDim.x + threadIdx.x;
    if (i < n) { oa[i] = a[i]; ob[i] = b[i]; }
}

// ---------------------------------------------------------------------------
// bf16 WMMA GEMM: C[M,N] = A[M,Kp] * B + bias row + Dmat.
// A: [M,Kp] row-major bf16, lda = KT*32 (padded, pad columns zero).
// B: fragment-packed (k_pack_b_frag), streamed directly from global/L2.
// Block = 256 threads (8 wave32). Block tile 64x128; wave tile 16x64.
// A-tile double-buffered in LDS via async global->LDS b128 + ASYNCcnt.
// ---------------------------------------------------------------------------
__global__ __launch_bounds__(256)
void k_gemm_bf16(const bf16_t* __restrict__ A, int lda,
                 const bf16_t* __restrict__ BP, int KT,
                 float* __restrict__ C, int ldc,
                 const float* __restrict__ bias,
                 const float* __restrict__ Dm, int ldd,
                 int M, int N) {
    __shared__ alignas(32) bf16_t As[2][64 * 32];   // double-buffered [m][k] tile

    const int tid  = threadIdx.x;
    const int lane = tid & 31;
    const int wave = tid >> 5;
    const int wm   = wave >> 1;                   // 0..3 wave row
    const int wn   = wave & 1;                    // 0..1 wave col
    const int m0   = blockIdx.y * 64;
    const int nt0  = blockIdx.x * 8 + wn * 4;     // first N-tile-of-16
    const int fm   = lane & 15;
    const int kh   = lane >> 4;

    // A staging: one 16B chunk per thread per tile; clamp row (dup rows are
    // discarded by the epilogue M-guard; WMMA raises no exceptions).
    const int ar = tid >> 2;                      // 0..63 tile row
    const int ac = (tid & 3) << 3;                // 0,8,16,24 k offset
    const int gm = m0 + ar;
    const int gmc = (gm < M) ? gm : (M - 1);
    const bf16_t* asrc = A + (long)gmc * lda + ac;

    const bf16_t* bp[4];
#pragma unroll
    for (int i = 0; i < 4; ++i)
        bp[i] = BP + ((long)(nt0 + i) * KT) * 512 + lane * 16;

    v8f acc[4] = {};

    auto stage = [&](int kt, int buf) {
#if HAVE_ASYNC_LDS
        __builtin_amdgcn_global_load_async_to_lds_b128(
            (gv4i*)(gvoid*)(asrc + ((long)kt << 5)),
            (lv4i*)(lvoid*)&As[buf][ar * 32 + ac], 0, 0);
#else
        const bf16_t* s = asrc + ((long)kt << 5);
        bf16_t* d = &As[buf][ar * 32 + ac];
#pragma unroll
        for (int j = 0; j < 8; ++j) d[j] = s[j];
#endif
    };
    auto compute = [&](int buf, int kt) {
        const v16bf a = *(const v16bf*)&As[buf][(wm * 16 + fm) * 32 + kh * 16];
#pragma unroll
        for (int i = 0; i < 4; ++i) {
            const v16bf b = *(const v16bf*)(bp[i] + (long)kt * 512);
            acc[i] = __builtin_amdgcn_wmma_f32_16x16x32_bf16(
                false, a, false, b, (short)0, acc[i], false, false);
        }
    };

    // Software pipeline: prefetch tile kt+1 while computing tile kt.
    stage(0, 0);
    for (int kt = 0; kt < KT - 1; ++kt) {
        const int cur = kt & 1;
        stage(kt + 1, cur ^ 1);
#pragma unroll
        for (int i = 0; i < 4; ++i)
            __builtin_prefetch(bp[i] + (long)(kt + 1) * 512, 0, 1);
#if HAVE_ASYNC_LDS
        __builtin_amdgcn_s_wait_asynccnt(1);   // in-order: tile kt is resident
#endif
        __syncthreads();
        compute(cur, kt);
        __syncthreads();
    }
#if HAVE_ASYNC_LDS
    __builtin_amdgcn_s_wait_asynccnt(0);
#endif
    __syncthreads();
    compute((KT - 1) & 1, KT - 1);

    // Epilogue. C/D frag: lanes 0-15 -> M=r, lanes 16-31 -> M=r+8; N=lane&15.
#pragma unroll
    for (int i = 0; i < 4; ++i) {
        int gn = (nt0 + i) * 16 + fm;
        if (gn >= N) continue;
        float bv = bias ? bias[gn] : 0.0f;
#pragma unroll
        for (int r = 0; r < 8; ++r) {
            int gmr = m0 + wm * 16 + r + kh * 8;
            if (gmr >= M) continue;
            float v = acc[i][r] + bv;
            if (Dm) v += Dm[(long)gmr * ldd + gn];
            C[(long)gmr * ldc + gn] = v;
        }
    }
}

// ---------------------------------------------------------------------------
// LSTM gate elementwise (PyTorch gate order i,f,g,o); writes padded bf16 h.
// ---------------------------------------------------------------------------
__global__ void k_lstm_gate(const float* __restrict__ g, int G, int ldp,
                            float* __restrict__ c,
                            float* __restrict__ hF,
                            bf16_t* __restrict__ hB,
                            bf16_t* __restrict__ hseq_t) {
    int i = blockIdx.x * blockDim.x + threadIdx.x;
    if (i >= B_SZ * G) return;
    int b = i / G, j = i % G;
    const float* gr = g + (long)b * 4 * G;
    float gi = gr[j], gf = gr[G + j], gg = gr[2 * G + j], go = gr[3 * G + j];
    float si = 1.0f / (1.0f + __expf(-gi));
    float sf = 1.0f / (1.0f + __expf(-gf));
    float tg = tanhf(gg);
    float so = 1.0f / (1.0f + __expf(-go));
    float cn = sf * c[i] + si * tg;
    float hn = so * tanhf(cn);
    c[i]  = cn;
    hF[i] = hn;
    hB[(long)b * ldp + j]     = (bf16_t)hn;
    hseq_t[(long)b * ldp + j] = (bf16_t)hn;
}

// ---------------------------------------------------------------------------
extern "C" void kernel_launch(void* const* d_in, const int* in_sizes, int n_in,
                              void* d_out, int out_size, void* d_ws, size_t ws_size,
                              hipStream_t stream) {
    (void)in_sizes; (void)n_in; (void)out_size; (void)ws_size;
    const int*   tok  = (const int*)  d_in[0];
    const float* s1h  = (const float*)d_in[1];
    const float* s1c  = (const float*)d_in[2];
    const float* s2h  = (const float*)d_in[3];
    const float* s2c  = (const float*)d_in[4];
    const float* s3h  = (const float*)d_in[5];
    const float* s3c  = (const float*)d_in[6];
    const float* Wemb = (const float*)d_in[7];
    const float* fcb  = (const float*)d_in[8];
    const float* Wih1 = (const float*)d_in[9];
    const float* Whh1 = (const float*)d_in[10];
    const float* bih1 = (const float*)d_in[11];
    const float* bhh1 = (const float*)d_in[12];
    const float* Wih2 = (const float*)d_in[13];
    const float* Whh2 = (const float*)d_in[14];
    const float* bih2 = (const float*)d_in[15];
    const float* bhh2 = (const float*)d_in[16];
    const float* Wih3 = (const float*)d_in[17];
    const float* Whh3 = (const float*)d_in[18];
    const float* bih3 = (const float*)d_in[19];
    const float* bhh3 = (const float*)d_in[20];
    float* out = (float*)d_out;

    char* ws = (char*)d_ws; size_t off = 0;
    auto alloc = [&](size_t bytes) -> void* {
        void* p = ws + off; off = (off + bytes + 255) & ~(size_t)255; return p;
    };
    auto cdiv = [](int a, int b) { return (a + b - 1) / b; };
    auto packed_elems = [&](int N, int K) -> size_t {
        return (size_t)cdiv(N, 128) * 8 * cdiv(K, 32) * 512;
    };

    bf16_t* x0b   = (bf16_t*)alloc((size_t)N_ROWS * EP_SZ * 2);
    bf16_t* WembP = (bf16_t*)alloc(packed_elems(V_SZ, E_SZ) * 2);
    bf16_t* W1iP  = (bf16_t*)alloc(packed_elems(4 * H_SZ, E_SZ) * 2);
    bf16_t* W1hP  = (bf16_t*)alloc(packed_elems(4 * H_SZ, H_SZ) * 2);
    bf16_t* W2iP  = (bf16_t*)alloc(packed_elems(4 * H_SZ, H_SZ) * 2);
    bf16_t* W2hP  = (bf16_t*)alloc(packed_elems(4 * H_SZ, H_SZ) * 2);
    bf16_t* W3iP  = (bf16_t*)alloc(packed_elems(4 * E_SZ, H_SZ) * 2);
    bf16_t* W3hP  = (bf16_t*)alloc(packed_elems(4 * E_SZ, E_SZ) * 2);
    float*  b1    = (float*)alloc((size_t)4 * H_SZ * 4);
    float*  b2    = (float*)alloc((size_t)4 * H_SZ * 4);
    float*  b3    = (float*)alloc((size_t)4 * E_SZ * 4);
    float*  xg    = (float*)alloc((size_t)N_ROWS * 4 * H_SZ * 4);
    float*  gbuf  = (float*)alloc((size_t)B_SZ * 4 * H_SZ * 4);
    bf16_t* h1s   = (bf16_t*)alloc((size_t)N_ROWS * HP_SZ * 2);
    bf16_t* h2s   = (bf16_t*)alloc((size_t)N_ROWS * HP_SZ * 2);
    bf16_t* h3s   = (bf16_t*)alloc((size_t)N_ROWS * EP_SZ * 2);
    float*  hF    = (float*)alloc((size_t)B_SZ * H_SZ * 4);
    float*  cst   = (float*)alloc((size_t)B_SZ * H_SZ * 4);
    bf16_t* hB    = (bf16_t*)alloc((size_t)B_SZ * HP_SZ * 2);

    auto packB = [&](const float* W, bf16_t* BP, int N, int K) {
        int KT = cdiv(K, 32);
        long tot = (long)cdiv(N, 128) * 8 * KT * 512;
        k_pack_b_frag<<<(int)((tot + 255) / 256), 256, 0, stream>>>(W, BP, N, K, KT, tot);
    };
    auto gemm = [&](const bf16_t* A, int lda, const bf16_t* BP,
                    float* C, int ldc, const float* bias,
                    const float* Dm, int ldd, int M, int Nn, int K) {
        dim3 grid(cdiv(Nn, 128), cdiv(M, 64));
        k_gemm_bf16<<<grid, 256, 0, stream>>>(A, lda, BP, cdiv(K, 32),
                                              C, ldc, bias, Dm, ldd, M, Nn);
    };

    // ---- weight packing into WMMA fragment layout ----
    packB(Wemb, WembP, V_SZ, E_SZ);
    packB(Wih1, W1iP, 4 * H_SZ, E_SZ);
    packB(Whh1, W1hP, 4 * H_SZ, H_SZ);
    packB(Wih2, W2iP, 4 * H_SZ, H_SZ);
    packB(Whh2, W2hP, 4 * H_SZ, H_SZ);
    packB(Wih3, W3iP, 4 * E_SZ, H_SZ);
    packB(Whh3, W3hP, 4 * E_SZ, E_SZ);
    k_bias_sum<<<cdiv(4 * H_SZ, 256), 256, 0, stream>>>(bih1, bhh1, b1, 4 * H_SZ);
    k_bias_sum<<<cdiv(4 * H_SZ, 256), 256, 0, stream>>>(bih2, bhh2, b2, 4 * H_SZ);
    k_bias_sum<<<cdiv(4 * E_SZ, 256), 256, 0, stream>>>(bih3, bhh3, b3, 4 * E_SZ);

    // ---- embedding gather + pad-zeroing of h sequence buffers ----
    {
        long tot = (long)N_ROWS * EP_SZ;
        k_embed<<<(int)((tot + 255) / 256), 256, 0, stream>>>(tok, Wemb, x0b);
    }
    k_zero_pad<<<cdiv(N_ROWS * (HP_SZ - H_SZ), 256), 256, 0, stream>>>(
        h1s, N_ROWS, H_SZ, HP_SZ);
    k_zero_pad<<<cdiv(N_ROWS * (HP_SZ - H_SZ), 256), 256, 0, stream>>>(
        h2s, N_ROWS, H_SZ, HP_SZ);
    k_zero_pad<<<cdiv(N_ROWS * (EP_SZ - E_SZ), 256), 256, 0, stream>>>(
        h3s, N_ROWS, E_SZ, EP_SZ);

    // ---- one LSTM layer: hoisted xg GEMM + sequential scan ----
    auto run_layer = [&](const bf16_t* xseq, int in_ld, int inDim,
                         const bf16_t* WiP, const bf16_t* WhP, const float* bsum,
                         const float* h0, const float* c0, int G, int ldp,
                         bf16_t* hseq, float* outH, float* outC) {
        // xg = xseq @ Wi^T + (bih+bhh) over all T*B rows at once
        gemm(xseq, in_ld, WiP, xg, 4 * G, bsum, nullptr, 0, N_ROWS, 4 * G, inDim);
        k_init_state<<<cdiv(B_SZ * ldp, 256), 256, 0, stream>>>(h0, c0, hF, hB, cst, G, ldp);
        int nst = B_SZ * G;
        for (int t = 0; t < T_SEQ; ++t) {
            // g = xg[t] + h_prev @ Wh^T (xg slice fused as accumulator addend)
            gemm(hB, ldp, WhP, gbuf, 4 * G, nullptr,
                 xg + (size_t)t * B_SZ * 4 * G, 4 * G, B_SZ, 4 * G, G);
            k_lstm_gate<<<cdiv(nst, 256), 256, 0, stream>>>(
                gbuf, G, ldp, cst, hF, hB, hseq + (size_t)t * B_SZ * ldp);
        }
        k_copy2<<<cdiv(nst, 256), 256, 0, stream>>>(hF, cst, outH, outC, nst);
    };

    const size_t SC = (size_t)N_ROWS * V_SZ;     // scores element count
    run_layer(x0b, EP_SZ, E_SZ, W1iP, W1hP, b1, s1h, s1c, H_SZ, HP_SZ, h1s,
              out + SC,          out + SC + 92000);
    run_layer(h1s, HP_SZ, H_SZ, W2iP, W2hP, b2, s2h, s2c, H_SZ, HP_SZ, h2s,
              out + SC + 184000, out + SC + 276000);
    run_layer(h2s, HP_SZ, H_SZ, W3iP, W3hP, b3, s3h, s3c, E_SZ, EP_SZ, h3s,
              out + SC + 368000, out + SC + 400000);

    // ---- tied output GEMM: scores = h3 @ Wemb^T + fc_b ----
    gemm(h3s, EP_SZ, WembP, out, V_SZ, fcb, nullptr, 0, N_ROWS, V_SZ, E_SZ);
}
